// Multi_MultiC_GramCluster_v2_16131897164375
// MI455X (gfx1250) — compile-verified, matrix operand order checked by reading, not run on hardware
//
#include <hip/hip_runtime.h>
#include <hip/hip_bf16.h>
#include <math.h>

typedef __attribute__((ext_vector_type(16))) _Float16 v16h;
typedef __attribute__((ext_vector_type(8)))  _Float16 v8h;
typedef __attribute__((ext_vector_type(8)))  float    v8f;
typedef __attribute__((ext_vector_type(4)))  int      v4i;

#define BB   64
#define CC   512
#define HWsz 784
#define DD   512
#define GRD  512
#define KCEN 20
#define NCL  10
#define LDSH 40          // LDS row stride in halfs: 80B, 16B-aligned sublanes
#define SPLIT 128        // K-split for gram_feat GEMM
#define KCHUNK (CC*CC/SPLIT)   // 2048

#define ACT_NONE   0
#define ACT_RELU   1
#define ACT_GELU   2
#define ACT_SIGMUL 3

#if __has_builtin(__builtin_amdgcn_global_load_async_to_lds_b128)
#define HAVE_ASYNC_B128 1
#else
#define HAVE_ASYNC_B128 0
#endif

#if HAVE_ASYNC_B128
typedef __attribute__((address_space(1))) v4i v4i_g;   // global int4
typedef __attribute__((address_space(3))) v4i v4i_l;   // LDS int4

__device__ __forceinline__ void async_copy_b128(const void* gsrc, void* ldst) {
  __builtin_amdgcn_global_load_async_to_lds_b128(
      (v4i_g*)gsrc, (v4i_l*)ldst, 0, 0);
}

__device__ __forceinline__ void wait_async0() {
#if __has_builtin(__builtin_amdgcn_s_wait_asynccnt)
  __builtin_amdgcn_s_wait_asynccnt(0);
#else
  asm volatile("s_wait_asynccnt 0x0" ::: "memory");
#endif
}
#endif

// ---- WMMA fragment loader: 16x32 f16 A/B fragment from LDS tile (row-major,
// stride LDSH halfs). lanes 0-15: row m, K {0..7,16..23}; lanes 16-31: K {8..15,24..31}
__device__ __forceinline__ v16h load_frag(const _Float16* s, int row, int half_id) {
  const _Float16* p = s + row * LDSH + half_id * 8;
  v8h lo = *(const v8h*)(p);        // K = half_id*8 .. +7
  v8h hi = *(const v8h*)(p + 16);   // K = 16 + half_id*8 .. +7
  union { v16h v; v8h h[2]; } u;
  u.h[0] = lo; u.h[1] = hi;
  return u.v;
}

__device__ __forceinline__ v8h cvt8(const float* p) {   // 16B-aligned fast path
  const float4 a0 = *(const float4*)(p);
  const float4 a1 = *(const float4*)(p + 4);
  v8h h;
  h[0]=(_Float16)a0.x; h[1]=(_Float16)a0.y; h[2]=(_Float16)a0.z; h[3]=(_Float16)a0.w;
  h[4]=(_Float16)a1.x; h[5]=(_Float16)a1.y; h[6]=(_Float16)a1.z; h[7]=(_Float16)a1.w;
  return h;
}

#define WMMA_F16(A, B, C) \
  __builtin_amdgcn_wmma_f32_16x16x32_f16(false, (A), false, (B), (short)0, (C), false, false)

// ============================================================================
// Gram: per batch b, G = (X Xt)/HW, X=[512,784] f32.  Output f16 [B,512,512].
// grid (8 colblk64, 4 rowblk128, 64 batch), 256 thr = 8 waves.
// Each wave: 32x32 register tile (2x2 WMMA) -> 8 ds_load_b128 per 4 WMMA.
// ============================================================================
__global__ void __launch_bounds__(256) gram_wmma_kernel(
    const float* __restrict__ ximg, _Float16* __restrict__ G16) {
  __shared__ _Float16 As[128 * LDSH];
  __shared__ _Float16 Bs[64 * LDSH];
  const int tj = blockIdx.x, ti = blockIdx.y, bz = blockIdx.z;
  const int tid = threadIdx.x;
  const int lane = tid & 31, wave = tid >> 5;
  const int half_id = lane >> 4, ml = lane & 15;
  const int i2 = wave & 3;     // row 32-block within 128 (0..3)
  const int j2 = wave >> 2;    // col 32-block within 64  (0..1)
  const float* Xb = ximg + (size_t)bz * CC * HWsz;

  // A staging: 128 rows x 32 halfs; thread: row 0..127, k-sub 0/16
  const int rA = tid >> 1, kA = (tid & 1) * 16;
  // B staging: 64 rows x 32 halfs; thread: row 0..63, k-sub 0/8/16/24
  const int rB = tid >> 2, kB = (tid & 3) * 8;

  v8f c00 = {}, c01 = {}, c10 = {}, c11 = {};

  for (int k0 = 0; k0 < HWsz; k0 += 32) {
    __syncthreads();
    const float* sA = Xb + (size_t)(ti * 128 + rA) * HWsz + k0 + kA;
    const float* sB = Xb + (size_t)(tj * 64 + rB) * HWsz + k0 + kB;
    if (k0 + 32 <= HWsz) {          // full tile: vector convert
      *(v8h*)(As + rA * LDSH + kA)     = cvt8(sA);
      *(v8h*)(As + rA * LDSH + kA + 8) = cvt8(sA + 8);
      *(v8h*)(Bs + rB * LDSH + kB)     = cvt8(sB);
    } else {                        // tail (k0=768): zero-pad K>=784
      v8h h0, h1, hb;
#pragma unroll
      for (int e = 0; e < 8; ++e) {
        h0[e] = (k0 + kA + e     < HWsz) ? (_Float16)sA[e]     : (_Float16)0.0f;
        h1[e] = (k0 + kA + 8 + e < HWsz) ? (_Float16)sA[8 + e] : (_Float16)0.0f;
        hb[e] = (k0 + kB + e     < HWsz) ? (_Float16)sB[e]     : (_Float16)0.0f;
      }
      *(v8h*)(As + rA * LDSH + kA)     = h0;
      *(v8h*)(As + rA * LDSH + kA + 8) = h1;
      *(v8h*)(Bs + rB * LDSH + kB)     = hb;
    }
    __syncthreads();

    const v16h a0 = load_frag(As, i2 * 32 + ml,      half_id);
    const v16h a1 = load_frag(As, i2 * 32 + 16 + ml, half_id);
    const v16h b0 = load_frag(Bs, j2 * 32 + ml,      half_id);
    const v16h b1 = load_frag(Bs, j2 * 32 + 16 + ml, half_id);
    c00 = WMMA_F16(a0, b0, c00);
    c01 = WMMA_F16(a0, b1, c01);
    c10 = WMMA_F16(a1, b0, c10);
    c11 = WMMA_F16(a1, b1, c11);
  }

  const float inv = 1.0f / (float)HWsz;
  _Float16* Gb = G16 + (size_t)bz * CC * CC;
  const int mbase = ti * 128 + i2 * 32;
  const int nbase = tj * 64 + j2 * 32 + ml;
#pragma unroll
  for (int v = 0; v < 8; ++v) {
    const int m0 = mbase + v + 8 * half_id;
    const int m1 = m0 + 16;
    Gb[(size_t)m0 * CC + nbase]      = (_Float16)(c00[v] * inv);
    Gb[(size_t)m0 * CC + nbase + 16] = (_Float16)(c01[v] * inv);
    Gb[(size_t)m1 * CC + nbase]      = (_Float16)(c10[v] * inv);
    Gb[(size_t)m1 * CC + nbase + 16] = (_Float16)(c11[v] * inv);
  }
}

// ============================================================================
// gram_feat partials: [64, 262144](f16) @ [262144, 512](f32->f16)
// grid (8 nblk, SPLIT ksplit); each WG: full M=64 x 64 cols x 2048 K-chunk.
// A slice (already f16) staged via GLOBAL_LOAD_ASYNC_TO_LDS when available.
// Memory-bound on gram_w (512 MB @ 23.3 TB/s ~= 22 us floor).
// ============================================================================
__global__ void __launch_bounds__(256) gramfeat_wmma_kernel(
    const _Float16* __restrict__ G16, const float* __restrict__ gram_w,
    float* __restrict__ partial) {
  __shared__ _Float16 As[64 * LDSH];
  __shared__ _Float16 Bs[64 * LDSH];   // transposed: Bs[n][k]
  const int nb = blockIdx.x;           // 0..7  (64-col block)
  const int sp = blockIdx.y;           // 0..SPLIT-1
  const int tid = threadIdx.x;
  const int lane = tid & 31, wave = tid >> 5;
  const int half_id = lane >> 4, ml = lane & 15;
  const int i = wave & 3, jp = wave >> 2;
  const int kbase = sp * KCHUNK;

  const int ra = tid >> 2, ka = (tid & 3) * 8;   // A staging: row 0..63, k off
  const int kb = tid >> 3, nn = (tid & 7) * 8;   // B staging: k 0..31, n off

  v8f acc0 = {}; v8f acc1 = {};

  for (int kc = 0; kc < KCHUNK; kc += 32) {
    const int k0 = kbase + kc;
    __syncthreads();
    // A slice: 64 batches x 32 k (already f16) -> LDS
#if HAVE_ASYNC_B128
    async_copy_b128(G16 + (size_t)ra * (CC * CC) + k0 + ka,
                    As + ra * LDSH + ka);
#else
    *(v8h*)(As + ra * LDSH + ka) =
        *(const v8h*)(G16 + (size_t)ra * (CC * CC) + k0 + ka);
#endif
    // B slice: gram_w[k0+kb][nb*64+nn .. +8] -> Bs[n][k] (f32->f16 transpose)
    {
      const float* src = gram_w + (size_t)(k0 + kb) * GRD + nb * 64 + nn;
      const float4 f0 = *(const float4*)(src);
      const float4 f1 = *(const float4*)(src + 4);
      Bs[(nn + 0) * LDSH + kb] = (_Float16)f0.x;
      Bs[(nn + 1) * LDSH + kb] = (_Float16)f0.y;
      Bs[(nn + 2) * LDSH + kb] = (_Float16)f0.z;
      Bs[(nn + 3) * LDSH + kb] = (_Float16)f0.w;
      Bs[(nn + 4) * LDSH + kb] = (_Float16)f1.x;
      Bs[(nn + 5) * LDSH + kb] = (_Float16)f1.y;
      Bs[(nn + 6) * LDSH + kb] = (_Float16)f1.z;
      Bs[(nn + 7) * LDSH + kb] = (_Float16)f1.w;
    }
#if HAVE_ASYNC_B128
    wait_async0();
#endif
    __syncthreads();

    const v16h a  = load_frag(As, i * 16 + ml, half_id);
    const v16h b0 = load_frag(Bs, (jp * 2 + 0) * 16 + ml, half_id);
    const v16h b1 = load_frag(Bs, (jp * 2 + 1) * 16 + ml, half_id);
    acc0 = WMMA_F16(a, b0, acc0);
    acc1 = WMMA_F16(a, b1, acc1);
  }

  float* pp = partial + (size_t)sp * (BB * GRD);
  const int n0 = nb * 64 + (jp * 2 + 0) * 16 + ml;
#pragma unroll
  for (int v = 0; v < 8; ++v) {
    const int m = i * 16 + v + 8 * half_id;
    pp[(size_t)m * GRD + n0]      = acc0[v];
    pp[(size_t)m * GRD + n0 + 16] = acc1[v];
  }
}

__global__ void __launch_bounds__(256) gram_reduce_kernel(
    const float* __restrict__ partial, const float* __restrict__ gram_b,
    float* __restrict__ gram_feat) {
  const int idx = blockIdx.x * 256 + threadIdx.x;   // 0..32767
  float acc = gram_b[idx & (GRD - 1)];
  for (int s = 0; s < SPLIT; ++s) acc += partial[(size_t)s * (BB * GRD) + idx];
  gram_feat[idx] = acc;
}

// ============================================================================
// Generic small row-GEMM: Y[m,n] = act(dot(X[m,:],W[:,n]) + bias[n]) (+res)
// grid (ceil(N/256), M).  X row staged in LDS; W reads coalesced (L2 resident).
// ============================================================================
__global__ void __launch_bounds__(256) gemm_row_kernel(
    const float* __restrict__ X, const float* __restrict__ W,
    const float* __restrict__ bias, const float* __restrict__ resid,
    const float* __restrict__ mulaux, float* __restrict__ Y,
    int K, int N, int act) {
  __shared__ float xs[2048];
  const int m = blockIdx.y, tid = threadIdx.x;
  for (int k = tid; k < K; k += 256) xs[k] = X[(size_t)m * K + k];
  __syncthreads();
  const int n = blockIdx.x * 256 + tid;
  if (n >= N) return;
  float acc = bias[n];
#pragma unroll 8
  for (int k = 0; k < K; ++k) acc = fmaf(xs[k], W[(size_t)k * N + n], acc);
  if (act == ACT_RELU)        acc = fmaxf(acc, 0.0f);
  else if (act == ACT_GELU)   acc = 0.5f * acc * (1.0f + erff(acc * 0.70710678118654752f));
  else if (act == ACT_SIGMUL) acc = mulaux[(size_t)m * N + n] / (1.0f + expf(-acc));
  if (resid) acc += resid[(size_t)m * N + n];
  Y[(size_t)m * N + n] = acc;
}

__global__ void __launch_bounds__(256) layernorm_kernel(
    const float* __restrict__ X, const float* __restrict__ g,
    const float* __restrict__ b, float* __restrict__ Y, int Dn) {
  __shared__ float r1[256], r2[256];
  const int m = blockIdx.x, tid = threadIdx.x;
  float s = 0.0f, s2 = 0.0f;
  for (int k = tid; k < Dn; k += 256) {
    const float v = X[(size_t)m * Dn + k];
    s += v; s2 += v * v;
  }
  r1[tid] = s; r2[tid] = s2;
  __syncthreads();
  for (int off = 128; off > 0; off >>= 1) {
    if (tid < off) { r1[tid] += r1[tid + off]; r2[tid] += r2[tid + off]; }
    __syncthreads();
  }
  const float mean = r1[0] / (float)Dn;
  const float var  = r2[0] / (float)Dn - mean * mean;
  const float inv  = rsqrtf(var + 1e-5f);
  for (int k = tid; k < Dn; k += 256)
    Y[(size_t)m * Dn + k] = (X[(size_t)m * Dn + k] - mean) * inv * g[k] + b[k];
}

// nearest centroid (squared euclid, lowest index on ties) -> x_fused[:,512:]
__global__ void __launch_bounds__(32) cluster_kernel(
    const float* __restrict__ gf, const float* __restrict__ centers,
    float* __restrict__ x_fused) {
  const int bz = blockIdx.x, lane = threadIdx.x;
  __shared__ float d2s[32];
  __shared__ int best;
  float d2 = 3.4e38f;
  if (lane < KCEN) {
    float acc = 0.0f;
    const float* c = centers + (size_t)lane * GRD;
    const float* x = gf + (size_t)bz * GRD;
    for (int j = 0; j < GRD; ++j) { const float d = x[j] - c[j]; acc = fmaf(d, d, acc); }
    d2 = acc;
  }
  d2s[lane] = d2;
  __syncthreads();
  if (lane == 0) {
    int bi = 0; float bv = d2s[0];
    for (int k = 1; k < KCEN; ++k) if (d2s[k] < bv) { bv = d2s[k]; bi = k; }
    best = bi;
  }
  __syncthreads();
  const float* c = centers + (size_t)best * GRD;
  for (int j = lane; j < GRD; j += 32)
    x_fused[(size_t)bz * (DD + GRD) + DD + j] = c[j];
}

__global__ void __launch_bounds__(256) concat_first_kernel(
    const float* __restrict__ x_main, float* __restrict__ x_fused) {
  const int idx = blockIdx.x * 256 + threadIdx.x;   // 0..32767
  const int m = idx >> 9, j = idx & 511;
  x_fused[(size_t)m * (DD + GRD) + j] = x_main[idx];
}

// ============================================================================
extern "C" void kernel_launch(void* const* d_in, const int* in_sizes, int n_in,
                              void* d_out, int out_size, void* d_ws, size_t ws_size,
                              hipStream_t stream) {
  const float* x_embed = (const float*)d_in[0];
  const float* x_image = (const float*)d_in[1];
  const float* attn_w1 = (const float*)d_in[2];
  const float* attn_b1 = (const float*)d_in[3];
  const float* attn_w2 = (const float*)d_in[4];
  const float* attn_b2 = (const float*)d_in[5];
  const float* ln_ff_g = (const float*)d_in[6];
  const float* ln_ff_b = (const float*)d_in[7];
  const float* ff_w1   = (const float*)d_in[8];
  const float* ff_b1   = (const float*)d_in[9];
  const float* ff_w2   = (const float*)d_in[10];
  const float* ff_b2   = (const float*)d_in[11];
  const float* gram_w  = (const float*)d_in[12];
  const float* gram_b  = (const float*)d_in[13];
  const float* centers = (const float*)d_in[14];
  const float* gm_w    = (const float*)d_in[15];
  const float* gm_b    = (const float*)d_in[16];
  const float* ln1_g   = (const float*)d_in[17];
  const float* ln1_b   = (const float*)d_in[18];
  const float* m1_w    = (const float*)d_in[19];
  const float* m1_b    = (const float*)d_in[20];
  const float* m2_w1   = (const float*)d_in[21];
  const float* m2_b1   = (const float*)d_in[22];
  const float* m2_w2   = (const float*)d_in[23];
  const float* m2_b2   = (const float*)d_in[24];
  const float* out_w   = (const float*)d_in[25];
  const float* out_b   = (const float*)d_in[26];
  float* logits = (float*)d_out;

  char* wsp = (char*)d_ws;
  size_t off = 0;
  auto alloc = [&](size_t bytes) -> void* {
    void* p = wsp + off;
    off += (bytes + 255) & ~(size_t)255;
    return p;
  };
  float* h1        = (float*)alloc((size_t)BB * DD * 4);
  float* x_attn    = (float*)alloc((size_t)BB * DD * 4);
  float* t0        = (float*)alloc((size_t)BB * DD * 4);
  float* hff       = (float*)alloc((size_t)BB * 4 * DD * 4);
  float* x_main    = (float*)alloc((size_t)BB * DD * 4);
  float* x_h       = (float*)alloc((size_t)BB * DD * 4);
  float* t1        = (float*)alloc((size_t)BB * DD * 4);
  float* hbuf      = (float*)alloc((size_t)BB * DD * 4);
  float* h2        = (float*)alloc((size_t)BB * DD * 4);
  float* h3        = (float*)alloc((size_t)BB * (DD / 2) * 4);
  float* gram_feat = (float*)alloc((size_t)BB * GRD * 4);
  float* x_fused   = (float*)alloc((size_t)BB * (DD + GRD) * 4);
  float* partial   = (float*)alloc((size_t)SPLIT * BB * GRD * 4);
  _Float16* G16    = (_Float16*)alloc((size_t)BB * CC * CC * 2);
  (void)ws_size; (void)in_sizes; (void)n_in; (void)out_size;

  const dim3 blk(256);

  // ---- gating MLP + residual FF branch (tiny, L2-resident) ----
  gemm_row_kernel<<<dim3(2, BB), blk, 0, stream>>>(x_embed, attn_w1, attn_b1, nullptr, nullptr, h1, DD, DD, ACT_RELU);
  gemm_row_kernel<<<dim3(2, BB), blk, 0, stream>>>(h1, attn_w2, attn_b2, nullptr, x_embed, x_attn, DD, DD, ACT_SIGMUL);
  layernorm_kernel<<<dim3(BB), blk, 0, stream>>>(x_attn, ln_ff_g, ln_ff_b, t0, DD);
  gemm_row_kernel<<<dim3(8, BB), blk, 0, stream>>>(t0, ff_w1, ff_b1, nullptr, nullptr, hff, DD, 4 * DD, ACT_GELU);
  gemm_row_kernel<<<dim3(2, BB), blk, 0, stream>>>(hff, ff_w2, ff_b2, x_attn, nullptr, x_main, 4 * DD, DD, ACT_NONE);

  // ---- Gram (WMMA) ----
  gram_wmma_kernel<<<dim3(8, 4, BB), blk, 0, stream>>>(x_image, G16);
  gramfeat_wmma_kernel<<<dim3(8, SPLIT), blk, 0, stream>>>(G16, gram_w, partial);
  gram_reduce_kernel<<<dim3(BB * GRD / 256), blk, 0, stream>>>(partial, gram_b, gram_feat);

  // ---- cluster lookup + fuse ----
  concat_first_kernel<<<dim3(BB * DD / 256), blk, 0, stream>>>(x_main, x_fused);
  cluster_kernel<<<dim3(BB), dim3(32), 0, stream>>>(gram_feat, centers, x_fused);

  // ---- head ----
  gemm_row_kernel<<<dim3(2, BB), blk, 0, stream>>>(x_fused, gm_w, gm_b, nullptr, nullptr, x_h, DD + GRD, DD, ACT_GELU);
  layernorm_kernel<<<dim3(BB), blk, 0, stream>>>(x_h, ln1_g, ln1_b, t1, DD);
  gemm_row_kernel<<<dim3(2, BB), blk, 0, stream>>>(t1, m1_w, m1_b, x_h, nullptr, hbuf, DD, DD, ACT_GELU);
  gemm_row_kernel<<<dim3(2, BB), blk, 0, stream>>>(hbuf, m2_w1, m2_b1, nullptr, nullptr, h2, DD, DD, ACT_GELU);
  gemm_row_kernel<<<dim3(1, BB), blk, 0, stream>>>(h2, m2_w2, m2_b2, nullptr, nullptr, h3, DD, DD / 2, ACT_GELU);
  gemm_row_kernel<<<dim3(1, BB), blk, 0, stream>>>(h3, out_w, out_b, nullptr, nullptr, logits, DD / 2, NCL, ACT_NONE);
}
